// ModernMultiHeadAttention_12661563588910
// MI455X (gfx1250) — matrix operand order repaired and checked by reading, not compiled, not run on hardware
//
#include <hip/hip_runtime.h>
#include <hip/hip_bf16.h>
#include <math.h>

// ---------------------------------------------------------------------------
// Problem constants (from reference: B=2, S=4096, D=512, H=8, hd=64)
// ---------------------------------------------------------------------------
#define BATCH   2
#define SEQ     4096
#define DMODEL  512
#define NHEADS  8
#define HDIM    64
#define SROW    (SEQ + 8)          // padded LDS row stride (floats)

typedef __attribute__((ext_vector_type(16))) __bf16 v16bf;
typedef __attribute__((ext_vector_type(8)))  float  v8f;

union FragB {
    v16bf v;
    unsigned u[8];
    uint4 q[2];
};

__device__ __forceinline__ unsigned short f2bf(float f) {
    unsigned u = __float_as_uint(f);
    u += 0x7FFFu + ((u >> 16) & 1u);          // round-to-nearest-even
    return (unsigned short)(u >> 16);
}

// A fragment (16x32 bf16): lane m = L&15, hi = L>>4.
// VGPR0..3 -> K = kbase + hi*8 + {0..7}; VGPR4..7 -> K = kbase + 16 + hi*8 + {0..7}
__device__ __forceinline__ void loadA(FragB& f, const unsigned short* row, int kbase, int hi) {
    const unsigned short* p = row + kbase + hi * 8;
    f.q[0] = *(const uint4*)(p);
    f.q[1] = *(const uint4*)(p + 16);
}
// B fragment (32x16 bf16): lane n = L&15, hi = L>>4.
// VGPR0..3 -> K = kbase + hi*8 + {0..7}; VGPR4..7 -> K = kbase + 16 + hi*8 + {0..7}
__device__ __forceinline__ void loadB(FragB& f, const unsigned short* row, int kbase, int hi) {
    const unsigned short* p = row + kbase + hi * 8;
    f.q[0] = *(const uint4*)(p);
    f.q[1] = *(const uint4*)(p + 16);
}

__device__ __forceinline__ v8f wmma_bf16(const FragB& a, const FragB& b, v8f c) {
    return __builtin_amdgcn_wmma_f32_16x16x32_bf16(false, a.v, false, b.v,
                                                   (short)0, c, false, false);
}

// ---------------------------------------------------------------------------
// Async global->LDS staging via inline CDNA5 asm (ASYNCcnt-tracked DMA).
// VDST = per-lane LDS byte offset (low 32 bits of generic shared pointer,
// per the flat-aperture rule); VADDR = 64-bit address.
// One contiguous 2KB K tile (16 rows x 64 bf16): 4 x 16B per lane.
// ---------------------------------------------------------------------------
__device__ __forceinline__ void prefetch_tile_async(const unsigned short* gsrc,
                                                    const unsigned short* ldst, int lane) {
#pragma unroll
    for (int i = 0; i < 4; ++i) {
        const void* g = gsrc + lane * 8 + i * 256;
        unsigned l = (unsigned)(unsigned long long)(ldst + lane * 8 + i * 256);
        asm volatile("global_load_async_to_lds_b128 %0, %1, off"
                     :: "v"(l), "v"(g) : "memory");
    }
}
__device__ __forceinline__ void wait_async_le4() {
    asm volatile("s_wait_asynccnt 4" ::: "memory");
}
__device__ __forceinline__ void wait_async_0() {
    asm volatile("s_wait_asynccnt 0" ::: "memory");
}

// ---------------------------------------------------------------------------
// fp32 -> bf16 bulk convert
// ---------------------------------------------------------------------------
__global__ __launch_bounds__(256) void conv_bf16_kernel(const float* __restrict__ src,
                                                        unsigned short* __restrict__ dst, int n) {
    int i = blockIdx.x * 256 + threadIdx.x;
    if (i < n) dst[i] = f2bf(src[i]);
}

// ---------------------------------------------------------------------------
// GEMM: Y[m,n] = sum_k A[m,k] * W[n,k]   (torch Linear: x @ W^T)
// One wave computes a 16x64 output tile: 4 accumulators share one A fragment
// per K-step (4x A reuse, 4 WMMAs per 32-K slice). 8 waves per block.
// ---------------------------------------------------------------------------
__global__ __launch_bounds__(256) void gemm_bf16_kernel(const unsigned short* __restrict__ A,
                                                        const unsigned short* __restrict__ W,
                                                        float* __restrict__ Y,
                                                        int M, int N, int Kd) {
    const int lane = threadIdx.x & 31;
    const int wave = threadIdx.x >> 5;
    const int tile = blockIdx.x * 8 + wave;           // over (M/16)*(N/64)
    const int ntn  = N >> 6;
    const int mt   = tile / ntn;
    const int nq   = tile - mt * ntn;                 // 64-wide column group
    const int ln   = lane & 15;
    const int hi   = lane >> 4;

    const unsigned short* arow = A + (size_t)(mt * 16 + ln) * Kd;
    const unsigned short* w0 = W + (size_t)(nq * 64 +  0 + ln) * Kd;
    const unsigned short* w1 = W + (size_t)(nq * 64 + 16 + ln) * Kd;
    const unsigned short* w2 = W + (size_t)(nq * 64 + 32 + ln) * Kd;
    const unsigned short* w3 = W + (size_t)(nq * 64 + 48 + ln) * Kd;

    v8f acc0 = {}, acc1 = {}, acc2 = {}, acc3 = {};
    for (int k = 0; k < Kd; k += 32) {
        FragB fa, f0, f1, f2, f3;
        loadA(fa, arow, k, hi);
        loadB(f0, w0, k, hi);
        loadB(f1, w1, k, hi);
        loadB(f2, w2, k, hi);
        loadB(f3, w3, k, hi);
        acc0 = wmma_bf16(fa, f0, acc0);
        acc1 = wmma_bf16(fa, f1, acc1);
        acc2 = wmma_bf16(fa, f2, acc2);
        acc3 = wmma_bf16(fa, f3, acc3);
    }
    float* ybase = Y + (size_t)(mt * 16) * N + nq * 64 + ln;
#pragma unroll
    for (int j = 0; j < 8; ++j) {
        const size_t row = (size_t)(j + hi * 8) * N;
        ybase[row +  0] = acc0[j];
        ybase[row + 16] = acc1[j];
        ybase[row + 32] = acc2[j];
        ybase[row + 48] = acc3[j];
    }
}

// ---------------------------------------------------------------------------
// RoPE on Q,K + bf16 convert to head-major [B,H,S,hd]; V converted and
// transposed to [B,H,hd,S] so PV WMMA B-fragments are contiguous per lane.
// ---------------------------------------------------------------------------
__global__ __launch_bounds__(256) void rope_convert_kernel(const float* __restrict__ Qf,
                                                           const float* __restrict__ Kf,
                                                           const float* __restrict__ Vf,
                                                           unsigned short* __restrict__ Qb,
                                                           unsigned short* __restrict__ Kb,
                                                           unsigned short* __restrict__ Vt) {
    const int t = blockIdx.x * 256 + threadIdx.x;      // 2^21 threads total
    if (t >= BATCH * NHEADS * SEQ * 32) return;
    const int i = t & 31;
    const int s = (t >> 5) & (SEQ - 1);
    const int h = (t >> 17) & (NHEADS - 1);
    const int b = t >> 20;

    const size_t tok   = (size_t)b * SEQ + s;
    const size_t ebase = tok * DMODEL + h * HDIM;
    const float inv_freq = powf(10000.0f, -((float)(2 * i)) / (float)HDIM);
    const float ang = (float)s * inv_freq;
    const float c = cosf(ang), sn = sinf(ang);

    const float q0 = Qf[ebase + i], q1 = Qf[ebase + i + 32];
    const float k0 = Kf[ebase + i], k1 = Kf[ebase + i + 32];

    const size_t obase = (((size_t)(b * NHEADS + h) * SEQ) + s) * HDIM;
    Qb[obase + i]      = f2bf(q0 * c - q1 * sn);
    Qb[obase + i + 32] = f2bf(q1 * c + q0 * sn);
    Kb[obase + i]      = f2bf(k0 * c - k1 * sn);
    Kb[obase + i + 32] = f2bf(k1 * c + k0 * sn);

    const size_t vb = (size_t)(b * NHEADS + h) * HDIM;
    Vt[(vb + i)      * SEQ + s] = f2bf(Vf[ebase + i]);
    Vt[(vb + i + 32) * SEQ + s] = f2bf(Vf[ebase + i + 32]);
}

// ---------------------------------------------------------------------------
// Fused attention: one workgroup = one (b, h, 16-query tile).
// Phase 1: S = Q K^T * scale via WMMA into 256KB LDS (16 x 4096 f32), with
//          K tiles double-buffered through LDS via async-to-LDS DMA.
// Phase 2: rowwise softmax; probs written once to HBM (f32) and packed bf16
//          in place into the LDS rows for phase 3.
// Phase 3: ctx = P V via WMMA; A frags are direct ds_load_b128 of bf16 probs.
// ---------------------------------------------------------------------------
__global__ __launch_bounds__(256) void attn_kernel(const unsigned short* __restrict__ Qb,
                                                   const unsigned short* __restrict__ Kb,
                                                   const unsigned short* __restrict__ Vt,
                                                   float* __restrict__ attn_out,
                                                   unsigned short* __restrict__ ctx) {
    extern __shared__ float lds[];
    float* partial = lds + 16 * SROW;
    __shared__ __attribute__((aligned(16))) unsigned short kstage[8][2][16 * HDIM]; // 32KB

    const int lane = threadIdx.x & 31;
    const int wave = threadIdx.x >> 5;
    const int bid  = blockIdx.x;
    const int qt   = bid & 255;
    const int h    = (bid >> 8) & (NHEADS - 1);
    const int b    = bid >> 11;
    const int ln   = lane & 15;
    const int hi   = lane >> 4;
    const size_t bh = (size_t)(b * NHEADS + h);
    const float scale = 0.125f;                  // hd^-0.5 = 1/8

    // --- Phase 1: scores (K tiles double-buffered via async DMA to LDS) ---
    const unsigned short* qrow = Qb + (bh * SEQ + qt * 16 + ln) * HDIM;
    FragB qa0, qa1;
    loadA(qa0, qrow, 0, hi);
    loadA(qa1, qrow, 32, hi);

    {
        const unsigned short* kbase = Kb + bh * SEQ * HDIM;
        const int ntiles = (SEQ / 16) / 8;                     // 32 tiles per wave
        prefetch_tile_async(kbase + (size_t)wave * 16 * HDIM, kstage[wave][0], lane);
        int cur = 0;
        for (int i = 0; i < ntiles; ++i) {
            const int kt = wave + 8 * i;
            if (i + 1 < ntiles) {
                prefetch_tile_async(kbase + (size_t)(kt + 8) * 16 * HDIM,
                                    kstage[wave][cur ^ 1], lane);
                wait_async_le4();                              // batch i landed
            } else {
                wait_async_0();
            }
            const unsigned short* krow = &kstage[wave][cur][ln * HDIM];
            FragB fk0, fk1;
            loadB(fk0, krow, 0, hi);
            loadB(fk1, krow, 32, hi);
            v8f acc = {};
            acc = wmma_bf16(qa0, fk0, acc);
            acc = wmma_bf16(qa1, fk1, acc);
#pragma unroll
            for (int j = 0; j < 8; ++j)
                lds[(j + hi * 8) * SROW + kt * 16 + ln] = acc[j] * scale;
            cur ^= 1;
        }
    }
    __syncthreads();

    // --- Phase 2: softmax, two rows per wave; bf16 probs packed in place ---
    for (int r = wave * 2; r < wave * 2 + 2; ++r) {
        float* srow = lds + r * SROW;
        unsigned short* brow = (unsigned short*)srow;
        float mx = -3.402823466e38f;
        for (int c = lane; c < SEQ; c += 32) mx = fmaxf(mx, srow[c]);
#pragma unroll
        for (int o = 16; o > 0; o >>= 1) mx = fmaxf(mx, __shfl_xor(mx, o, 32));
        float sum = 0.0f;
        for (int c = lane; c < SEQ; c += 32) {
            float e = __expf(srow[c] - mx);
            srow[c] = e;
            sum += e;
        }
#pragma unroll
        for (int o = 16; o > 0; o >>= 1) sum += __shfl_xor(sum, o, 32);
        const float inv = 1.0f / sum;
        float* gout = attn_out + ((bh * SEQ) + qt * 16 + r) * (size_t)SEQ;
        for (int c = lane; c < SEQ; c += 32) {
            float p = srow[c] * inv;
            gout[c] = p;             // mandatory streaming 1GB attn_weights write
            brow[c] = f2bf(p);       // bf16 in place; clobbers only consumed f32s
        }
    }
    __syncthreads();

    // --- Phase 3: ctx = P V ---
    const int ntile = wave & 3;                   // output column tile (hd = 4x16)
    const int half  = wave >> 2;                  // split keys across wave halves
    const unsigned short* vrow = Vt + (bh * HDIM + ntile * 16 + ln) * (size_t)SEQ;
    const unsigned short* prow = (const unsigned short*)(lds + ln * SROW);

    v8f acc = {};
    const int k0 = half * (SEQ / 2);
    for (int kb = k0; kb < k0 + SEQ / 2; kb += 32) {
        FragB fa, fv;
        loadA(fa, prow, kb, hi);                  // bf16 probs via ds_load_b128
        loadB(fv, vrow, kb, hi);
        acc = wmma_bf16(fa, fv, acc);
    }
    float* pw = partial + wave * 256;
#pragma unroll
    for (int j = 0; j < 8; ++j) pw[(j + hi * 8) * 16 + ln] = acc[j];
    __syncthreads();

    if (wave < 4) {
        const float* p0 = partial + wave * 256;
        const float* p1 = partial + (wave + 4) * 256;
#pragma unroll
        for (int j = 0; j < 8; ++j) {
            const int row = j + hi * 8;
            const float v = p0[row * 16 + ln] + p1[row * 16 + ln];
            ctx[((size_t)b * SEQ + qt * 16 + row) * DMODEL + h * HDIM + wave * 16 + ln] = f2bf(v);
        }
    }
}

// ---------------------------------------------------------------------------
// Host side
// ---------------------------------------------------------------------------
extern "C" void kernel_launch(void* const* d_in, const int* in_sizes, int n_in,
                              void* d_out, int out_size, void* d_ws, size_t ws_size,
                              hipStream_t stream) {
    (void)in_sizes; (void)n_in; (void)out_size; (void)ws_size;

    const float* x  = (const float*)d_in[0];
    const float* Wq = (const float*)d_in[1];
    const float* Wk = (const float*)d_in[2];
    const float* Wv = (const float*)d_in[3];
    const float* Wo = (const float*)d_in[4];

    float* out_ptr  = (float*)d_out;                                   // [B,S,D]
    float* attn_ptr = out_ptr + (size_t)BATCH * SEQ * DMODEL;          // [B,H,S,S]

    const size_t nTok = (size_t)BATCH * SEQ;        // 8192
    const size_t nX   = nTok * DMODEL;              // 4,194,304
    const size_t nW   = (size_t)DMODEL * DMODEL;    // 262,144

    char* p = (char*)d_ws;
    unsigned short* xb  = (unsigned short*)p;  p += nX * 2;
    unsigned short* Wqb = (unsigned short*)p;  p += nW * 2;
    unsigned short* Wkb = (unsigned short*)p;  p += nW * 2;
    unsigned short* Wvb = (unsigned short*)p;  p += nW * 2;
    unsigned short* Wob = (unsigned short*)p;  p += nW * 2;
    float* Qf = (float*)p;  p += nX * 4;
    float* Kf = (float*)p;  p += nX * 4;
    float* Vf = (float*)p;  p += nX * 4;
    unsigned short* Qb  = (unsigned short*)p;  p += nX * 2;
    unsigned short* Kb  = (unsigned short*)p;  p += nX * 2;
    unsigned short* Vt  = (unsigned short*)p;  p += nX * 2;
    unsigned short* ctx = (unsigned short*)p;  p += nX * 2;

    // 1) fp32 -> bf16 conversions
    conv_bf16_kernel<<<(int)((nX + 255) / 256), 256, 0, stream>>>(x,  xb,  (int)nX);
    conv_bf16_kernel<<<(int)((nW + 255) / 256), 256, 0, stream>>>(Wq, Wqb, (int)nW);
    conv_bf16_kernel<<<(int)((nW + 255) / 256), 256, 0, stream>>>(Wk, Wkb, (int)nW);
    conv_bf16_kernel<<<(int)((nW + 255) / 256), 256, 0, stream>>>(Wv, Wvb, (int)nW);
    conv_bf16_kernel<<<(int)((nW + 255) / 256), 256, 0, stream>>>(Wo, Wob, (int)nW);

    // 2) QKV projections: (8192/16)*(512/64) = 4096 wave-tiles / 8 = 512 blocks
    const int gemmBlocks = (int)((nTok / 16) * (DMODEL / 64) / 8);
    gemm_bf16_kernel<<<gemmBlocks, 256, 0, stream>>>(xb, Wqb, Qf, (int)nTok, DMODEL, DMODEL);
    gemm_bf16_kernel<<<gemmBlocks, 256, 0, stream>>>(xb, Wkb, Kf, (int)nTok, DMODEL, DMODEL);
    gemm_bf16_kernel<<<gemmBlocks, 256, 0, stream>>>(xb, Wvb, Vf, (int)nTok, DMODEL, DMODEL);

    // 3) RoPE + bf16 + V transpose
    const int ropeThreads = BATCH * NHEADS * SEQ * 32;
    rope_convert_kernel<<<ropeThreads / 256, 256, 0, stream>>>(Qf, Kf, Vf, Qb, Kb, Vt);

    // 4) fused attention: 256KB scores + 8KB partials dynamic + 32KB static LDS
    const int ldsBytes = (16 * SROW + 8 * 256) * (int)sizeof(float);   // 270,848 B
    (void)hipFuncSetAttribute((const void*)attn_kernel,
                              hipFuncAttributeMaxDynamicSharedMemorySize, ldsBytes);
    const int attnBlocks = BATCH * NHEADS * (SEQ / 16);                // 4096
    attn_kernel<<<attnBlocks, 256, ldsBytes, stream>>>(Qb, Kb, Vt, attn_ptr, ctx);

    // 5) output projection -> d_out
    gemm_bf16_kernel<<<gemmBlocks, 256, 0, stream>>>(ctx, Wob, out_ptr, (int)nTok, DMODEL, DMODEL);
}
